// MultiHeadMaskedAttention_2637109919991
// MI455X (gfx1250) — matrix operand order repaired and checked by reading, not compiled, not run on hardware
//
#include <hip/hip_runtime.h>
#include <hip/hip_bf16.h>

// ---------------------------------------------------------------------------
// MI455X (gfx1250) multi-head causal attention.
// Matmuls: v_wmma_f32_16x16x32_f16 (wave32, f32 accumulate).
// Tile staging: global_load_async_to_lds_b128 (ASYNCcnt) where it's a pure
// copy; 128-bit load/convert/store where conversion or transpose is needed.
// ---------------------------------------------------------------------------

typedef __attribute__((ext_vector_type(16))) _Float16 v16h;
typedef __attribute__((ext_vector_type(8)))  float    v8f;
typedef __attribute__((ext_vector_type(4)))  float    f4;
typedef __attribute__((ext_vector_type(4)))  _Float16 h4;
typedef __attribute__((ext_vector_type(8)))  _Float16 h8;
typedef __attribute__((ext_vector_type(4)))  unsigned u4;

union HFrag { unsigned u[8]; v16h h; };

constexpr int D_MODEL   = 1024;
constexpr int NUM_HEADS = 16;
constexpr int DH        = 64;
constexpr int BATCH     = 2;
constexpr int SEQ       = 2048;
constexpr int MTOT      = BATCH * SEQ;   // 4096 rows

__device__ __forceinline__ v8f wmma_f16(const HFrag& a, const HFrag& b, v8f c) {
  return __builtin_amdgcn_wmma_f32_16x16x32_f16(false, a.h, false, b.h,
                                                (short)0, c, false, false);
}

// Async global -> LDS copy, 16 bytes per lane, tracked by ASYNCcnt.
// lds_off: byte offset within LDS (low 32 bits of a generic pointer to
// __shared__ memory on AMDGCN). g: 64-bit global address.
__device__ __forceinline__ void async_copy_b128(const _Float16* lds_dst,
                                                const _Float16* g) {
  unsigned lds_off = (unsigned)(unsigned long long)lds_dst;
  asm volatile("global_load_async_to_lds_b128 %0, %1, off"
               :: "v"(lds_off), "v"(g) : "memory");
}
__device__ __forceinline__ void wait_async() {
  asm volatile("s_wait_asynccnt 0x0" ::: "memory");
}

// ---------------------------------------------------------------------------
// GEMM:  out[M,N] = A[M,K] * W[K,N] + bias[N]
// MODE 0: A = fp32; (W,b,out) picked by blockIdx.z; out = f16 head-major.
// MODE 1: A = f16 row-major (async-staged); out = fp32 [M,N].
// Block tile 128x256, BK=64, 256 threads = 8 waves, wave tile 64x64.
// ---------------------------------------------------------------------------
constexpr int BM  = 128;
constexpr int BN  = 256;
constexpr int BK  = 64;
constexpr int LDT = BK + 8;   // LDS pitch in halves; row stride 144B (16B-aligned)

template <int MODE>
__global__ __launch_bounds__(256)
void gemm_kernel(const void* __restrict__ Av,
                 const float* __restrict__ Wq, const float* __restrict__ Wk,
                 const float* __restrict__ Wv,
                 const float* __restrict__ Bq, const float* __restrict__ Bk,
                 const float* __restrict__ Bv,
                 void* Oq, void* Ok, void* Ov) {
  __shared__ _Float16 ldsA[BM * LDT];   // [m][k]             18.4 KB
  __shared__ _Float16 ldsB[BN * LDT];   // transposed [n][k]  36.9 KB

  const int tid  = threadIdx.x;
  const int lane = tid & 31;
  const int wave = tid >> 5;
  const int half = lane >> 4;
  const int ln   = lane & 15;
  const int wm   = wave >> 2;   // 0..1 : 64-row group
  const int wn   = wave & 3;    // 0..3 : 64-col group
  const int n0   = blockIdx.x * BN;
  const int m0   = blockIdx.y * BM;

  const float* W;  const float* bias;  void* Optr;
  if (MODE == 0) {
    if (blockIdx.z == 0)      { W = Wq; bias = Bq; Optr = Oq; }
    else if (blockIdx.z == 1) { W = Wk; bias = Bk; Optr = Ok; }
    else                      { W = Wv; bias = Bv; Optr = Ov; }
  } else {
    W = Wq; bias = Bq; Optr = Oq;
  }

  v8f acc[4][4];
#pragma unroll
  for (int mt = 0; mt < 4; ++mt)
#pragma unroll
    for (int nt = 0; nt < 4; ++nt) acc[mt][nt] = v8f{};

  for (int k0 = 0; k0 < D_MODEL; k0 += BK) {
    __syncthreads();

    // prefetch next k-step tiles into L2 while this stage computes
    if (k0 + BK < D_MODEL) {
      __builtin_prefetch((const char*)W +
          ((size_t)(k0 + BK + (tid >> 2)) * D_MODEL + n0 + (tid & 3) * 64) * 4, 0, 1);
      if (MODE == 0)
        __builtin_prefetch((const float*)Av +
            (size_t)(m0 + (tid >> 1)) * D_MODEL + k0 + BK + (tid & 1) * 32, 0, 1);
    }

    // ---- stage A tile (128 x 64) as f16 ----
    if (MODE == 0) {
      const float* A = (const float*)Av;
#pragma unroll
      for (int i = 0; i < 8; ++i) {                 // 2048 float4 / 256 thr
        int f = tid + i * 256, e = f * 4, r = e >> 6, c = e & 63;
        f4 v = *(const f4*)&A[(size_t)(m0 + r) * D_MODEL + k0 + c];
        h4 hv; hv[0] = (_Float16)v[0]; hv[1] = (_Float16)v[1];
        hv[2] = (_Float16)v[2]; hv[3] = (_Float16)v[3];
        *(h4*)&ldsA[r * LDT + c] = hv;
      }
    } else {
      const _Float16* A = (const _Float16*)Av;
#pragma unroll
      for (int i = 0; i < 4; ++i) {                 // 1024 x16B async copies
        int f = tid + i * 256, e = f * 8, r = e >> 6, c = e & 63;
        async_copy_b128(&ldsA[r * LDT + c],
                        &A[(size_t)(m0 + r) * D_MODEL + k0 + c]);
      }
    }
    // ---- stage W tile (64 x 256) transposed: ldsB[n][k] ----
#pragma unroll
    for (int i = 0; i < 16; ++i) {                  // 4096 float4 / 256 thr
      int f = tid + i * 256, e = f * 4, r = e >> 8, c = e & 255; // r=k, c=n
      f4 v = *(const f4*)&W[(size_t)(k0 + r) * D_MODEL + n0 + c];
#pragma unroll
      for (int j = 0; j < 4; ++j)
        ldsB[(c + j) * LDT + r] = (_Float16)v[j];
    }
    if (MODE == 1) wait_async();
    __syncthreads();

#pragma unroll
    for (int ks = 0; ks < BK / 32; ++ks) {
      const int ko = ks * 32;
      HFrag a[4];
#pragma unroll
      for (int mt = 0; mt < 4; ++mt) {
        const _Float16* ap = &ldsA[(wm * 64 + mt * 16 + ln) * LDT + ko];
#pragma unroll
        for (int i = 0; i < 8; ++i) {
          int kk = half * 16 + i * 2;
          a[mt].u[i] = *(const unsigned*)(ap + kk);
        }
      }
#pragma unroll
      for (int nt = 0; nt < 4; ++nt) {
        HFrag b;
        const _Float16* bp = &ldsB[(wn * 64 + nt * 16 + ln) * LDT + ko + half * 16];
#pragma unroll
        for (int i = 0; i < 8; ++i) b.u[i] = *(const unsigned*)(bp + 2 * i);
#pragma unroll
        for (int mt = 0; mt < 4; ++mt)
          acc[mt][nt] = wmma_f16(a[mt], b, acc[mt][nt]);
      }
    }
  }

  // ---- epilogue: C/D layout  m = half*8 + r, n = ln ----
#pragma unroll
  for (int mt = 0; mt < 4; ++mt)
#pragma unroll
    for (int nt = 0; nt < 4; ++nt)
#pragma unroll
      for (int r = 0; r < 8; ++r) {
        int m = m0 + wm * 64 + mt * 16 + half * 8 + r;
        int n = n0 + wn * 64 + nt * 16 + ln;
        float v = acc[mt][nt][r] + bias[n];
        if (MODE == 0) {
          int bb = m / SEQ, t = m % SEQ;
          int hh = n / DH,  d = n % DH;
          ((_Float16*)Optr)[(((size_t)bb * NUM_HEADS + hh) * SEQ + t) * DH + d] =
              (_Float16)v;
        } else {
          ((float*)Optr)[(size_t)m * D_MODEL + n] = v;
        }
      }
}

// ---------------------------------------------------------------------------
// Flash attention (causal). One workgroup per (b, h, 128 query rows).
// 8 waves x 16 query rows each; key blocks of 64; Dh = 64.
// ---------------------------------------------------------------------------
constexpr int QP = 72;   // pitch for 64-wide f16 rows (144B, 16B-aligned)

__global__ __launch_bounds__(256)
void attn_kernel(const _Float16* __restrict__ Q, const _Float16* __restrict__ K,
                 const _Float16* __restrict__ V, _Float16* __restrict__ Y) {
  __shared__ _Float16 ldsQK[128 * QP];      // Q staging (128 rows) -> K tiles (64 rows)
  __shared__ _Float16 ldsVt[64 * QP];       // V transposed: [d][key]
  __shared__ _Float16 ldsP[8 * 16 * QP];    // per-wave P tile 16 x 64

  const int tid  = threadIdx.x;
  const int lane = tid & 31;
  const int wave = tid >> 5;
  const int half = lane >> 4;
  const int ln   = lane & 15;
  const int qi   = blockIdx.x;      // query tile (128 rows)
  const int h    = blockIdx.y;
  const int b    = blockIdx.z;

  const size_t headOff = ((size_t)b * NUM_HEADS + h) * SEQ * DH;
  const _Float16* Qh = Q + headOff + (size_t)qi * 128 * DH;

  // stage Q tile 128x64 via async global->LDS (no VGPR round-trip)
#pragma unroll
  for (int i = 0; i < 4; ++i) {                    // 1024 x16B async copies
    int f = tid + i * 256, e = f * 8, r = e >> 6, c = e & 63;
    async_copy_b128(&ldsQK[r * QP + c], &Qh[r * DH + c]);
  }
  wait_async();
  __syncthreads();

  // cache Q A-fragments in registers (reused for every key block)
  HFrag aq[2];
#pragma unroll
  for (int ks = 0; ks < 2; ++ks) {
    const _Float16* ap = &ldsQK[(wave * 16 + ln) * QP + ks * 32];
#pragma unroll
    for (int i = 0; i < 8; ++i) {
      int kk = half * 16 + i * 2;
      aq[ks].u[i] = *(const unsigned*)(ap + kk);
    }
  }

  v8f oacc[4];
#pragma unroll
  for (int dt = 0; dt < 4; ++dt) oacc[dt] = v8f{};
  float run_max[8], run_sum[8];
#pragma unroll
  for (int r = 0; r < 8; ++r) { run_max[r] = -3.0e38f; run_sum[r] = 0.0f; }

  const float scale = 0.125f;                 // 1/sqrt(64)
  const int   nblk  = 2 * (qi + 1);           // causal: keys < (qi+1)*128

  for (int j = 0; j < nblk; ++j) {
    const _Float16* Kh = K + headOff + (size_t)j * 64 * DH;
    const _Float16* Vh = V + headOff + (size_t)j * 64 * DH;

    // prefetch next key block into L2
    if (j + 1 < nblk) {
      __builtin_prefetch(Kh + 64 * DH + tid * 16, 0, 1);
      __builtin_prefetch(Vh + 64 * DH + tid * 16, 0, 1);
    }

    __syncthreads();                          // everyone done with prev K / Vt
#pragma unroll
    for (int i = 0; i < 2; ++i) {             // K tile: 512 x16B async copies
      int f = tid + i * 256, e = f * 8, r = e >> 6, c = e & 63;
      async_copy_b128(&ldsQK[r * QP + c], &Kh[r * DH + c]);     // K [key][d]
      h8 v = *(const h8*)&Vh[r * DH + c];                       // V row chunk
#pragma unroll
      for (int jj = 0; jj < 8; ++jj)
        ldsVt[(c + jj) * QP + r] = v[jj];                       // Vt [d][key]
    }
    wait_async();
    __syncthreads();

    // ---- S = Q K^T : 4 column tiles of 16 keys ----
    v8f s[4];
#pragma unroll
    for (int nt = 0; nt < 4; ++nt) {
      v8f a = v8f{};
#pragma unroll
      for (int ks = 0; ks < 2; ++ks) {
        HFrag bk;
        const _Float16* bp = &ldsQK[(nt * 16 + ln) * QP + ks * 32 + half * 16];
#pragma unroll
        for (int i = 0; i < 8; ++i) bk.u[i] = *(const unsigned*)(bp + 2 * i);
        a = wmma_f16(aq[ks], bk, a);
      }
      s[nt] = a;
    }

    // ---- scale + causal mask ----
    const int qbase = qi * 128 + wave * 16 + half * 8;
#pragma unroll
    for (int nt = 0; nt < 4; ++nt) {
      int kc = j * 64 + nt * 16 + ln;
#pragma unroll
      for (int r = 0; r < 8; ++r) {
        float v = s[nt][r] * scale;
        if (kc > qbase + r) v = -3.0e38f;
        s[nt][r] = v;
      }
    }

    // ---- online softmax: each row lives in one 16-lane half-wave ----
#pragma unroll
    for (int r = 0; r < 8; ++r) {
      float mloc = s[0][r];
#pragma unroll
      for (int nt = 1; nt < 4; ++nt) mloc = fmaxf(mloc, s[nt][r]);
#pragma unroll
      for (int o = 1; o < 16; o <<= 1) mloc = fmaxf(mloc, __shfl_xor(mloc, o, 32));
      float mnew = fmaxf(run_max[r], mloc);
      float corr = __expf(run_max[r] - mnew);
      run_max[r] = mnew;
      float lsum = 0.0f;
#pragma unroll
      for (int nt = 0; nt < 4; ++nt) {
        float p = __expf(s[nt][r] - mnew);
        s[nt][r] = p;
        lsum += p;
      }
#pragma unroll
      for (int o = 1; o < 16; o <<= 1) lsum += __shfl_xor(lsum, o, 32);
      run_sum[r] = run_sum[r] * corr + lsum;
#pragma unroll
      for (int dt = 0; dt < 4; ++dt) oacc[dt][r] *= corr;
    }

    // ---- P -> wave-private LDS (reshape C-layout -> A-layout) ----
    _Float16* pw = ldsP + wave * 16 * QP;
#pragma unroll
    for (int nt = 0; nt < 4; ++nt)
#pragma unroll
      for (int r = 0; r < 8; ++r)
        pw[(half * 8 + r) * QP + nt * 16 + ln] = (_Float16)s[nt][r];

    // ---- O += P V ----
    HFrag ap[2];
#pragma unroll
    for (int ks = 0; ks < 2; ++ks) {
      const _Float16* app = pw + ln * QP + ks * 32;
#pragma unroll
      for (int i = 0; i < 8; ++i) {
        int kk = half * 16 + i * 2;
        ap[ks].u[i] = *(const unsigned*)(app + kk);
      }
    }
#pragma unroll
    for (int dt = 0; dt < 4; ++dt) {
#pragma unroll
      for (int ks = 0; ks < 2; ++ks) {
        HFrag bv;
        const _Float16* bp = &ldsVt[(dt * 16 + ln) * QP + ks * 32 + half * 16];
#pragma unroll
        for (int i = 0; i < 8; ++i) bv.u[i] = *(const unsigned*)(bp + 2 * i);
        oacc[dt] = wmma_f16(ap[ks], bv, oacc[dt]);
      }
    }
  }

  // ---- normalize + write Y as f16 [B,T,C] (row-major for out-proj) ----
#pragma unroll
  for (int dt = 0; dt < 4; ++dt)
#pragma unroll
    for (int r = 0; r < 8; ++r) {
      int t = qi * 128 + wave * 16 + half * 8 + r;
      int c = h * DH + dt * 16 + ln;
      float ov = oacc[dt][r] / run_sum[r];
      Y[((size_t)b * SEQ + t) * D_MODEL + c] = (_Float16)ov;
    }
}

// ---------------------------------------------------------------------------
extern "C" void kernel_launch(void* const* d_in, const int* in_sizes, int n_in,
                              void* d_out, int out_size, void* d_ws, size_t ws_size,
                              hipStream_t stream) {
  const float* x  = (const float*)d_in[0];
  const float* wq = (const float*)d_in[1];
  const float* bq = (const float*)d_in[2];
  const float* wk = (const float*)d_in[3];
  const float* bk = (const float*)d_in[4];
  const float* wv = (const float*)d_in[5];
  const float* bv = (const float*)d_in[6];
  const float* wo = (const float*)d_in[7];
  const float* bo = (const float*)d_in[8];

  const size_t elems = (size_t)MTOT * D_MODEL;   // 4M
  _Float16* qbuf = (_Float16*)d_ws;
  _Float16* kbuf = qbuf + elems;
  _Float16* vbuf = kbuf + elems;
  _Float16* ybuf = vbuf + elems;                 // 32 MiB total scratch

  // 1) fused QKV projection (z = 0,1,2 -> q,k,v), f16 head-major out
  dim3 g1(D_MODEL / BN, MTOT / BM, 3);           // (4, 32, 3)
  gemm_kernel<0><<<g1, 256, 0, stream>>>(x, wq, wk, wv, bq, bk, bv,
                                         qbuf, kbuf, vbuf);

  // 2) causal flash attention per (b, h, 128-row query tile)
  dim3 g2(SEQ / 128, NUM_HEADS, BATCH);          // (16, 16, 2)
  attn_kernel<<<g2, 256, 0, stream>>>(qbuf, kbuf, vbuf, ybuf);

  // 3) output projection: d_out = ybuf @ wo + bo  (fp32)
  dim3 g3(D_MODEL / BN, MTOT / BM, 1);           // (4, 32, 1)
  gemm_kernel<1><<<g3, 256, 0, stream>>>(ybuf, wo, nullptr, nullptr,
                                         bo, nullptr, nullptr,
                                         d_out, nullptr, nullptr);
}
